// DiffusionTransformerBlock_29171417874764
// MI455X (gfx1250) — matrix operand order repaired and checked by reading, not compiled, not run on hardware
//
#include <hip/hip_runtime.h>

// ---------------- types / constants ----------------
typedef __attribute__((ext_vector_type(16))) __bf16 bf16x16;
typedef __attribute__((ext_vector_type(8)))  __bf16 bf16x8;
typedef __attribute__((ext_vector_type(8)))  float  f32x8;
typedef __attribute__((ext_vector_type(4)))  float  f32x4;

#define N_TOK 3072
#define CA 128
#define CS 384
#define CZ 16
#define NH 4
#define HD 32
#define SPLITS 4
#define KPS (N_TOK / SPLITS)   // keys per split = 768
#define QSCALE 0.17677669529663687f  // 1/sqrt(32)

__device__ __forceinline__ f32x8 zero8() {
  f32x8 z = {0.f,0.f,0.f,0.f,0.f,0.f,0.f,0.f};
  return z;
}

__device__ __forceinline__ f32x8 wmma_bf16(bf16x16 a, bf16x16 b, f32x8 c) {
  return __builtin_amdgcn_wmma_f32_16x16x32_bf16(false, a, false, b, (short)0, c, false, false);
}

// A-matrix (16x32, MxK) load from row-major bf16 [row][col], ld in elements.
// Lane half h: elements 0..7 = K 8h+0..7 ; elements 8..15 = K 16+8h+0..7 ; row = lane%16.
__device__ __forceinline__ bf16x16 load_a16(const __bf16* src, int ld, int row0, int col0, int lane) {
  int hf = lane >> 4, m = lane & 15;
  const __bf16* r = src + (size_t)(row0 + m) * ld + col0;
  bf16x8 lo = *(const bf16x8*)(r + hf * 8);
  bf16x8 hi = *(const bf16x8*)(r + 16 + hf * 8);
  bf16x16 out;
#pragma unroll
  for (int j = 0; j < 8; ++j) { out[j] = lo[j]; out[8 + j] = hi[j]; }
  return out;
}

// Same A pattern but source is fp32 (convert to bf16).
__device__ __forceinline__ bf16x16 load_a16_f32(const float* src, int ld, int row0, int col0, int lane) {
  int hf = lane >> 4, m = lane & 15;
  const float* r = src + (size_t)(row0 + m) * ld + col0;
  bf16x16 out;
#pragma unroll
  for (int j = 0; j < 8; ++j) {
    out[j]     = (__bf16)r[hf * 8 + j];
    out[8 + j] = (__bf16)r[16 + hf * 8 + j];
  }
  return out;
}

// B-matrix (32x16, KxN) load from row-major B^T, i.e. srcT[n][k], ld in elements.
// Lane half h: elements 0..7 = K 8h+0..7 ; elements 8..15 = K 16+8h+0..7 ; column n = lane%16.
__device__ __forceinline__ bf16x16 load_b16(const __bf16* srcT, int ld, int row0, int col0, int lane) {
  int hf = lane >> 4, n = lane & 15;
  const __bf16* r = srcT + (size_t)(row0 + n) * ld + col0;
  bf16x8 lo = *(const bf16x8*)(r + hf * 8);
  bf16x8 hi = *(const bf16x8*)(r + 16 + hf * 8);
  bf16x16 out;
#pragma unroll
  for (int j = 0; j < 8; ++j) { out[j] = lo[j]; out[8 + j] = hi[j]; }
  return out;
}

// ---------------- K0: weight convert + transpose (fp32 KxN -> bf16 NxK) ----------------
__global__ void k_prep(const float* __restrict__ src, __bf16* __restrict__ dstT, int K, int Nc) {
  int i = blockIdx.x * 256 + threadIdx.x;
  if (i < K * Nc) {
    int k = i / Nc, n = i % Nc;
    dstT[(size_t)n * K + k] = (__bf16)src[i];
  }
}

// ---------------- K1: AdaLN1 + Q,K,V,G projections ----------------
__global__ __launch_bounds__(256) void k_adaln_qkvg(
    const float* __restrict__ a, const float* __restrict__ s, const float* __restrict__ s_w,
    const __bf16* __restrict__ scale_wT, const float* __restrict__ scale_b,
    const __bf16* __restrict__ shift_wT,
    const __bf16* __restrict__ wqT, const float* __restrict__ bq,
    const __bf16* __restrict__ wkT, const __bf16* __restrict__ wvT, const __bf16* __restrict__ wgT,
    __bf16* __restrict__ q_bf, __bf16* __restrict__ k_bf, __bf16* __restrict__ vT_bf,
    float* __restrict__ g_buf) {
  __shared__ __align__(32) __bf16 sn_lds[16][CS];
  __shared__ __align__(32) float  an_lds[16][CA];
  __shared__ __align__(32) __bf16 h_lds[16][CA];

  int tid = threadIdx.x, lane = tid & 31, w = tid >> 5;
  int hf = lane >> 4, c0 = lane & 15;
  int row0 = blockIdx.x * 16;

  // ---- LN(s)*s_w and LN(a): wave w handles rows 2w (lanes 0-15), 2w+1 (lanes 16-31)
  int r = 2 * w + hf;
  const float* srow = s + (size_t)(row0 + r) * CS;
  float sum = 0.f, sq = 0.f;
  for (int j = 0; j < 24; ++j) { float x = srow[c0 + 16 * j]; sum += x; sq += x * x; }
  for (int mk = 1; mk < 16; mk <<= 1) { sum += __shfl_xor(sum, mk, 32); sq += __shfl_xor(sq, mk, 32); }
  float mean = sum * (1.f / CS);
  float rs = rsqrtf(sq * (1.f / CS) - mean * mean + 1e-5f);
  for (int j = 0; j < 24; ++j) {
    int c = c0 + 16 * j;
    sn_lds[r][c] = (__bf16)((srow[c] - mean) * rs * s_w[c]);
  }
  const float* arow = a + (size_t)(row0 + r) * CA;
  sum = 0.f; sq = 0.f;
  for (int j = 0; j < 8; ++j) { float x = arow[c0 + 16 * j]; sum += x; sq += x * x; }
  for (int mk = 1; mk < 16; mk <<= 1) { sum += __shfl_xor(sum, mk, 32); sq += __shfl_xor(sq, mk, 32); }
  mean = sum * (1.f / CA);
  rs = rsqrtf(sq * (1.f / CA) - mean * mean + 1e-5f);
  for (int j = 0; j < 8; ++j) { int c = c0 + 16 * j; an_lds[r][c] = (arow[c] - mean) * rs; }
  __syncthreads();

  // ---- adaln matmuls: wave w -> output columns [16w, 16w+16)
  f32x8 accS = zero8(), accH = zero8();
  for (int kc = 0; kc < CS; kc += 32) {
    bf16x16 A = load_a16(&sn_lds[0][0], CS, 0, kc, lane);
    accS = wmma_bf16(A, load_b16(scale_wT, CS, 16 * w, kc, lane), accS);
    accH = wmma_bf16(A, load_b16(shift_wT, CS, 16 * w, kc, lane), accH);
  }
  int n = 16 * w + c0;
  float sb = scale_b[n];
#pragma unroll
  for (int v = 0; v < 8; ++v) {
    int m = v + 8 * hf;
    float hval = an_lds[m][n] / (1.f + __expf(-(accS[v] + sb))) + accH[v];
    h_lds[m][n] = (__bf16)hval;
  }
  __syncthreads();

  // ---- q,k,v,g projections
  f32x8 aq = zero8(), ak = zero8(), av = zero8(), ag = zero8();
  for (int kc = 0; kc < CA; kc += 32) {
    bf16x16 A = load_a16(&h_lds[0][0], CA, 0, kc, lane);
    aq = wmma_bf16(A, load_b16(wqT, CA, 16 * w, kc, lane), aq);
    ak = wmma_bf16(A, load_b16(wkT, CA, 16 * w, kc, lane), ak);
    av = wmma_bf16(A, load_b16(wvT, CA, 16 * w, kc, lane), av);
    ag = wmma_bf16(A, load_b16(wgT, CA, 16 * w, kc, lane), ag);
  }
  float bqv = bq[n];
#pragma unroll
  for (int v = 0; v < 8; ++v) {
    int m = v + 8 * hf;
    size_t gr = (size_t)(row0 + m);
    q_bf[gr * CA + n] = (__bf16)((aq[v] + bqv) * QSCALE);
    k_bf[gr * CA + n] = (__bf16)ak[v];
    vT_bf[(size_t)n * N_TOK + gr] = (__bf16)av[v];            // V transposed: [channel][token]
    g_buf[gr * CA + n] = 1.f / (1.f + __expf(-ag[v]));
  }
}

// ---------------- K2: split-K flash attention with fused z pair-bias ----------------
// grid = (192 q-blocks, SPLITS key-segments), block = 128 (wave = head).
// Partials: part_o[(qb*NH+head)*SPLITS+sp][16][32], part_ml[..][0..15]=m, [16..31]=l.
__global__ __launch_bounds__(128) void k_attn(
    const __bf16* __restrict__ q_bf, const __bf16* __restrict__ k_bf,
    const __bf16* __restrict__ vT_bf,
    const float* __restrict__ z, const float* __restrict__ ln_z_w,
    const float* __restrict__ ln_z_b, const float* __restrict__ wb,
    float* __restrict__ part_o, float* __restrict__ part_ml) {
  __shared__ float bias_lds[NH][16][32];
  __shared__ float p_lds[NH][16][32];
  __shared__ float wbw[CZ][NH];
  __shared__ float bconst[NH];

  int tid = threadIdx.x, lane = tid & 31, head = tid >> 5;
  int hf = lane >> 4, nl = lane & 15;
  int q0 = blockIdx.x * 16;
  int kbeg = blockIdx.y * KPS, kend = kbeg + KPS;

  if (tid < CZ * NH) { int c = tid >> 2, h = tid & 3; wbw[c][h] = ln_z_w[c] * wb[c * NH + h]; }
  if (tid < NH) {
    float acc = 0.f;
    for (int c = 0; c < CZ; ++c) acc += ln_z_b[c] * wb[c * NH + tid];
    bconst[tid] = acc;
  }
  __syncthreads();

  bf16x16 QA = load_a16(q_bf, CA, q0, head * HD, lane);  // Q pre-scaled by 1/sqrt(D)
  f32x8 acc0 = zero8(), acc1 = zero8();
  float mrun[8], lrun[8];
#pragma unroll
  for (int v = 0; v < 8; ++v) { mrun[v] = -1e30f; lrun[v] = 0.f; }

#pragma unroll 1
  for (int k0 = kbeg; k0 < kend; k0 += 32) {
    __syncthreads();  // previous bias tile fully consumed
    // cooperative: bias tile = (LN(z)*w + b) @ wb for all 4 heads.
    // z is a 604MB single-use stream -> non-temporal loads so q/k/vT stay L2-resident.
#pragma unroll 1
    for (int i = 0; i < 4; ++i) {
      int idx = i * 128 + tid;            // 0..511 pairs
      int m = idx >> 5, n = idx & 31;
      const f32x4* zp4 = (const f32x4*)(z + ((size_t)(q0 + m) * N_TOK + (size_t)(k0 + n)) * CZ);
      f32x4 z4[4];
#pragma unroll
      for (int t = 0; t < 4; ++t) z4[t] = __builtin_nontemporal_load(zp4 + t);
      float zv[CZ]; float ssum = 0.f;
#pragma unroll
      for (int c = 0; c < CZ; ++c) { zv[c] = z4[c >> 2][c & 3]; ssum += zv[c]; }
      float mean = ssum * (1.f / CZ); float d2 = 0.f;
#pragma unroll
      for (int c = 0; c < CZ; ++c) { float d = zv[c] - mean; d2 += d * d; }
      float rs = rsqrtf(d2 * (1.f / CZ) + 1e-5f);
      float b0 = bconst[0], b1 = bconst[1], b2 = bconst[2], b3 = bconst[3];
#pragma unroll
      for (int c = 0; c < CZ; ++c) {
        float zn = (zv[c] - mean) * rs;
        b0 += zn * wbw[c][0]; b1 += zn * wbw[c][1]; b2 += zn * wbw[c][2]; b3 += zn * wbw[c][3];
      }
      bias_lds[0][m][n] = b0; bias_lds[1][m][n] = b1;
      bias_lds[2][m][n] = b2; bias_lds[3][m][n] = b3;
    }
    __syncthreads();

    // S = Q K^T + bias  (bias enters as WMMA C operand)
    f32x8 c0v, c1v;
#pragma unroll
    for (int v = 0; v < 8; ++v) {
      int m = v + 8 * hf;
      c0v[v] = bias_lds[head][m][nl];
      c1v[v] = bias_lds[head][m][nl + 16];
    }
    f32x8 S0 = wmma_bf16(QA, load_b16(k_bf, CA, k0,      head * HD, lane), c0v);
    f32x8 S1 = wmma_bf16(QA, load_b16(k_bf, CA, k0 + 16, head * HD, lane), c1v);

    // online softmax (row = M = v + 8*hf; N across 16-lane group)
#pragma unroll
    for (int v = 0; v < 8; ++v) {
      float mx = fmaxf(S0[v], S1[v]);
      for (int mk = 1; mk < 16; mk <<= 1) mx = fmaxf(mx, __shfl_xor(mx, mk, 32));
      float mnew = fmaxf(mrun[v], mx);
      float sc = __expf(mrun[v] - mnew);
      float p0 = __expf(S0[v] - mnew), p1 = __expf(S1[v] - mnew);
      float rsum = p0 + p1;
      for (int mk = 1; mk < 16; mk <<= 1) rsum += __shfl_xor(rsum, mk, 32);
      lrun[v] = lrun[v] * sc + rsum;
      mrun[v] = mnew;
      acc0[v] *= sc; acc1[v] *= sc;
      int m = v + 8 * hf;
      p_lds[head][m][nl] = p0;
      p_lds[head][m][nl + 16] = p1;
    }
    // P (C layout) -> A layout via wave-private LDS, then O += P @ V
    bf16x16 PA = load_a16_f32(&p_lds[head][0][0], 32, 0, 0, lane);
    acc0 = wmma_bf16(PA, load_b16(vT_bf, N_TOK, head * HD,      k0, lane), acc0);
    acc1 = wmma_bf16(PA, load_b16(vT_bf, N_TOK, head * HD + 16, k0, lane), acc1);
  }

  // write split partials
  size_t pidx = (size_t)((blockIdx.x * NH + head) * SPLITS + blockIdx.y);
  float* po = part_o + pidx * (16 * 32);
  float* pm = part_ml + pidx * 32;
#pragma unroll
  for (int v = 0; v < 8; ++v) {
    int m = v + 8 * hf;
    po[m * 32 + nl]      = acc0[v];
    po[m * 32 + 16 + nl] = acc1[v];
    if (nl == 0) { pm[m] = mrun[v]; pm[16 + m] = lrun[v]; }
  }
}

// ---------------- K2b: combine split partials, apply gate, emit bf16 ----------------
__global__ __launch_bounds__(256) void k_attn_combine(
    const float* __restrict__ part_o, const float* __restrict__ part_ml,
    const float* __restrict__ g_buf, __bf16* __restrict__ ao_bf) {
  int qb = blockIdx.x, q0 = qb * 16;
#pragma unroll 1
  for (int e = threadIdx.x; e < NH * 16 * HD; e += 256) {   // 2048 elements
    int head = e >> 9;
    int rm = (e >> 5) & 15;
    int d = e & 31;
    size_t base = (size_t)(qb * NH + head) * SPLITS;
    float mmax = -1e30f;
#pragma unroll
    for (int sp = 0; sp < SPLITS; ++sp)
      mmax = fmaxf(mmax, part_ml[(base + sp) * 32 + rm]);
    float osum = 0.f, lsum = 0.f;
#pragma unroll
    for (int sp = 0; sp < SPLITS; ++sp) {
      float wgt = __expf(part_ml[(base + sp) * 32 + rm] - mmax);
      lsum += wgt * part_ml[(base + sp) * 32 + 16 + rm];
      osum += wgt * part_o[(base + sp) * (16 * 32) + rm * 32 + d];
    }
    size_t gr = (size_t)(q0 + rm);
    float g = g_buf[gr * CA + head * HD + d];
    ao_bf[gr * CA + head * HD + d] = (__bf16)(osum / lsum * g);
  }
}

// ---------------- K3: out-proj + gate + residual + AdaLN2 + SwiGLU FFN + final ----------------
__global__ __launch_bounds__(256) void k_ffn(
    const float* __restrict__ a, const float* __restrict__ s, const __bf16* __restrict__ ao_bf,
    const __bf16* __restrict__ woT, const __bf16* __restrict__ sg1T, const float* __restrict__ sg1_b,
    const float* __restrict__ aln2_s_w, const __bf16* __restrict__ scale2T,
    const float* __restrict__ scale2_b, const __bf16* __restrict__ shift2T,
    const __bf16* __restrict__ w1T, const __bf16* __restrict__ w2T, const __bf16* __restrict__ woutT,
    const __bf16* __restrict__ sg2T, const float* __restrict__ sg2_b,
    float* __restrict__ out) {
  __shared__ __align__(32) __bf16 sraw[16][CS];
  __shared__ __align__(32) __bf16 sn2[16][CS];
  __shared__ __align__(32) float  ares[16][CA];
  __shared__ __align__(32) float  an2[16][CA];
  __shared__ __align__(32) __bf16 h2[16][CA];
  __shared__ __align__(32) __bf16 gact[16][2 * CA];

  int tid = threadIdx.x, lane = tid & 31, w = tid >> 5;
  int hf = lane >> 4, c0 = lane & 15;
  int row0 = blockIdx.x * 16;
  int r = 2 * w + hf;

  // s: raw bf16 (for style gates) + LN(s)*aln2_s_w
  const float* srow = s + (size_t)(row0 + r) * CS;
  float sum = 0.f, sq = 0.f;
  for (int j = 0; j < 24; ++j) { float x = srow[c0 + 16 * j]; sum += x; sq += x * x; }
  for (int mk = 1; mk < 16; mk <<= 1) { sum += __shfl_xor(sum, mk, 32); sq += __shfl_xor(sq, mk, 32); }
  float mean = sum * (1.f / CS);
  float rs = rsqrtf(sq * (1.f / CS) - mean * mean + 1e-5f);
  for (int j = 0; j < 24; ++j) {
    int c = c0 + 16 * j;
    float x = srow[c];
    sraw[r][c] = (__bf16)x;
    sn2[r][c] = (__bf16)((x - mean) * rs * aln2_s_w[c]);
  }
  __syncthreads();

  // attn = (g*o) @ wo ; sgate1 = sigmoid(s@sgate1_w + b) ; attn_out = sgate1*attn + a
  f32x8 accA = zero8(), accG = zero8();
  for (int kc = 0; kc < CA; kc += 32)
    accA = wmma_bf16(load_a16(ao_bf, CA, row0, kc, lane), load_b16(woT, CA, 16 * w, kc, lane), accA);
  for (int kc = 0; kc < CS; kc += 32)
    accG = wmma_bf16(load_a16(&sraw[0][0], CS, 0, kc, lane), load_b16(sg1T, CS, 16 * w, kc, lane), accG);
  int n = 16 * w + c0;
  float gb = sg1_b[n];
#pragma unroll
  for (int v = 0; v < 8; ++v) {
    int m = v + 8 * hf;
    float at = accA[v] / (1.f + __expf(-(accG[v] + gb)));
    ares[m][n] = at + a[(size_t)(row0 + m) * CA + n];
  }
  __syncthreads();

  // LN(attn_out)
  {
    const float* arow = &ares[r][0];
    float su = 0.f, s2 = 0.f;
    for (int j = 0; j < 8; ++j) { float x = arow[c0 + 16 * j]; su += x; s2 += x * x; }
    for (int mk = 1; mk < 16; mk <<= 1) { su += __shfl_xor(su, mk, 32); s2 += __shfl_xor(s2, mk, 32); }
    float mn = su * (1.f / CA);
    float r2 = rsqrtf(s2 * (1.f / CA) - mn * mn + 1e-5f);
    for (int j = 0; j < 8; ++j) { int c = c0 + 16 * j; an2[r][c] = (arow[c] - mn) * r2; }
  }
  __syncthreads();

  // AdaLN2 -> h2
  f32x8 aS = zero8(), aH = zero8();
  for (int kc = 0; kc < CS; kc += 32) {
    bf16x16 A = load_a16(&sn2[0][0], CS, 0, kc, lane);
    aS = wmma_bf16(A, load_b16(scale2T, CS, 16 * w, kc, lane), aS);
    aH = wmma_bf16(A, load_b16(shift2T, CS, 16 * w, kc, lane), aH);
  }
  float sb = scale2_b[n];
#pragma unroll
  for (int v = 0; v < 8; ++v) {
    int m = v + 8 * hf;
    h2[m][n] = (__bf16)(an2[m][n] / (1.f + __expf(-(aS[v] + sb))) + aH[v]);
  }
  __syncthreads();

  // SwiGLU: gated = silu(h2@w1) * (h2@w2)  (wave w -> 16-col tiles w and w+8 of 256)
  for (int t = 0; t < 2; ++t) {
    int nt = 16 * (w + 8 * t);
    f32x8 b1 = zero8(), b2 = zero8();
    for (int kc = 0; kc < CA; kc += 32) {
      bf16x16 A = load_a16(&h2[0][0], CA, 0, kc, lane);
      b1 = wmma_bf16(A, load_b16(w1T, CA, nt, kc, lane), b1);
      b2 = wmma_bf16(A, load_b16(w2T, CA, nt, kc, lane), b2);
    }
#pragma unroll
    for (int v = 0; v < 8; ++v) {
      int m = v + 8 * hf;
      float x = b1[v];
      gact[m][nt + c0] = (__bf16)(x / (1.f + __expf(-x)) * b2[v]);
    }
  }
  __syncthreads();

  // ff = sigmoid(s@sgate2_w + b) * (gated@wout) ; out = ff + attn_out
  f32x8 aF = zero8(), aG2 = zero8();
  for (int kc = 0; kc < 2 * CA; kc += 32)
    aF = wmma_bf16(load_a16(&gact[0][0], 2 * CA, 0, kc, lane),
                   load_b16(woutT, 2 * CA, 16 * w, kc, lane), aF);
  for (int kc = 0; kc < CS; kc += 32)
    aG2 = wmma_bf16(load_a16(&sraw[0][0], CS, 0, kc, lane),
                    load_b16(sg2T, CS, 16 * w, kc, lane), aG2);
  float g2b = sg2_b[n];
#pragma unroll
  for (int v = 0; v < 8; ++v) {
    int m = v + 8 * hf;
    out[(size_t)(row0 + m) * CA + n] = aF[v] / (1.f + __expf(-(aG2[v] + g2b))) + ares[m][n];
  }
}

// ---------------- host launch ----------------
extern "C" void kernel_launch(void* const* d_in, const int* in_sizes, int n_in,
                              void* d_out, int out_size, void* d_ws, size_t ws_size,
                              hipStream_t stream) {
  const float* a            = (const float*)d_in[0];
  const float* s            = (const float*)d_in[1];
  const float* z            = (const float*)d_in[2];
  const float* aln1_s_w     = (const float*)d_in[3];
  const float* aln1_scale_w = (const float*)d_in[4];
  const float* aln1_scale_b = (const float*)d_in[5];
  const float* aln1_shift_w = (const float*)d_in[6];
  const float* wq           = (const float*)d_in[7];
  const float* bq           = (const float*)d_in[8];
  const float* wk           = (const float*)d_in[9];
  const float* wv           = (const float*)d_in[10];
  const float* ln_z_w       = (const float*)d_in[11];
  const float* ln_z_b       = (const float*)d_in[12];
  const float* wb           = (const float*)d_in[13];
  const float* wg           = (const float*)d_in[14];
  const float* wo           = (const float*)d_in[15];
  const float* sgate1_w     = (const float*)d_in[16];
  const float* sgate1_b     = (const float*)d_in[17];
  const float* aln2_s_w     = (const float*)d_in[18];
  const float* aln2_scale_w = (const float*)d_in[19];
  const float* aln2_scale_b = (const float*)d_in[20];
  const float* aln2_shift_w = (const float*)d_in[21];
  const float* w1           = (const float*)d_in[22];
  const float* w2           = (const float*)d_in[23];
  const float* wout         = (const float*)d_in[24];
  const float* sgate2_w     = (const float*)d_in[25];
  const float* sgate2_b     = (const float*)d_in[26];

  char* ws = (char*)d_ws;
  size_t off = 0;
  auto alloc = [&](size_t bytes) -> char* {
    char* p = ws + off;
    off += (bytes + 255) & ~(size_t)255;
    return p;
  };

  __bf16* aln1_scale_wT = (__bf16*)alloc(CS * CA * 2);
  __bf16* aln1_shift_wT = (__bf16*)alloc(CS * CA * 2);
  __bf16* wqT           = (__bf16*)alloc(CA * CA * 2);
  __bf16* wkT           = (__bf16*)alloc(CA * CA * 2);
  __bf16* wvT           = (__bf16*)alloc(CA * CA * 2);
  __bf16* wgT           = (__bf16*)alloc(CA * CA * 2);
  __bf16* woT           = (__bf16*)alloc(CA * CA * 2);
  __bf16* sgate1T       = (__bf16*)alloc(CS * CA * 2);
  __bf16* aln2_scale_wT = (__bf16*)alloc(CS * CA * 2);
  __bf16* aln2_shift_wT = (__bf16*)alloc(CS * CA * 2);
  __bf16* w1T           = (__bf16*)alloc(CA * 2 * CA * 2);
  __bf16* w2T           = (__bf16*)alloc(CA * 2 * CA * 2);
  __bf16* woutT         = (__bf16*)alloc(2 * CA * CA * 2);
  __bf16* sgate2T       = (__bf16*)alloc(CS * CA * 2);

  __bf16* q_bf  = (__bf16*)alloc((size_t)N_TOK * CA * 2);
  __bf16* k_bf  = (__bf16*)alloc((size_t)N_TOK * CA * 2);
  __bf16* vT_bf = (__bf16*)alloc((size_t)N_TOK * CA * 2);
  __bf16* ao_bf = (__bf16*)alloc((size_t)N_TOK * CA * 2);
  float*  g_buf = (float*) alloc((size_t)N_TOK * CA * 4);

  int nblk = N_TOK / 16;  // 192
  float* part_o  = (float*)alloc((size_t)nblk * NH * SPLITS * 16 * 32 * 4);
  float* part_ml = (float*)alloc((size_t)nblk * NH * SPLITS * 32 * 4);

  auto prep = [&](const float* src, __bf16* dstT, int K, int Nc) {
    int total = K * Nc;
    k_prep<<<(total + 255) / 256, 256, 0, stream>>>(src, dstT, K, Nc);
  };
  prep(aln1_scale_w, aln1_scale_wT, CS, CA);
  prep(aln1_shift_w, aln1_shift_wT, CS, CA);
  prep(wq, wqT, CA, CA);
  prep(wk, wkT, CA, CA);
  prep(wv, wvT, CA, CA);
  prep(wg, wgT, CA, CA);
  prep(wo, woT, CA, CA);
  prep(sgate1_w, sgate1T, CS, CA);
  prep(aln2_scale_w, aln2_scale_wT, CS, CA);
  prep(aln2_shift_w, aln2_shift_wT, CS, CA);
  prep(w1, w1T, CA, 2 * CA);
  prep(w2, w2T, CA, 2 * CA);
  prep(wout, woutT, 2 * CA, CA);
  prep(sgate2_w, sgate2T, CS, CA);

  k_adaln_qkvg<<<nblk, 256, 0, stream>>>(
      a, s, aln1_s_w, aln1_scale_wT, aln1_scale_b, aln1_shift_wT,
      wqT, bq, wkT, wvT, wgT, q_bf, k_bf, vT_bf, g_buf);

  dim3 agrid(nblk, SPLITS);
  k_attn<<<agrid, 128, 0, stream>>>(
      q_bf, k_bf, vT_bf, z, ln_z_w, ln_z_b, wb, part_o, part_ml);

  k_attn_combine<<<nblk, 256, 0, stream>>>(part_o, part_ml, g_buf, ao_bf);

  k_ffn<<<nblk, 256, 0, stream>>>(
      a, s, ao_bf, woT, sgate1T, sgate1_b,
      aln2_s_w, aln2_scale_wT, aln2_scale_b, aln2_shift_wT,
      w1T, w2T, woutT, sgate2T, sgate2_b,
      (float*)d_out);
}
